// LongConvolution_10101763080276
// MI455X (gfx1250) — matrix operand order repaired and checked by reading, not compiled, not run on hardware
//
#include <hip/hip_runtime.h>

// LongConvolution on MI455X (gfx1250):
//   out[b,t,h] = x[b,t,h] + sum_{s<=t} x[b,s,h] * filt[h, t-s]
// Block-Toeplitz GEMM on v_wmma_f32_16x16x32_f16:
//  - For fixed lag delta, A[i,k]=filt[h,delta+i-k] is t-invariant, so one A
//    fragment feeds NC=8 accumulators (t-span 512/wave) and WMMA N-columns
//    pack (4 batches x 4 t-subtiles) -> 100% matrix-pipe utilization.
//  - x lives in a 1024-slot circular f16 LDS buffer (slot = s & 1023); each
//    lag step stages only 32 fresh s-values.
//  - 2-region software pipeline: region `it` stores regs loaded in region
//    `it-1`, issues loads for `it+2` (no wait), then runs 8 WMMAs -> L2
//    latency hidden behind matrix work; ONE barrier per region (the store
//    slots are provably disjoint from the concurrent read window).
// ~68.7 GFLOP of f16 MACs; 144 MB working set is L2-resident (192 MB L2).

typedef __attribute__((ext_vector_type(16))) _Float16 v16h;
typedef __attribute__((ext_vector_type(8)))  float    v8f;

#define B_       4
#define L_       4096
#define H_       1024
#define HT       16                 // channels per workgroup (one per wave)
#define TG       64                 // t per C-tile (4 x 16 M-tiles in N)
#define NC       8                  // C-tiles per wave -> t-span 512
#define TSPAN    (NC * TG)          // 512
#define KW       32                 // WMMA K depth (f16) = chunk width
#define SBUF     1024               // circular s-buffer slots (pow2)
#define SMASK    (SBUF - 1)
#define FW       64                 // filter window (padded pow2): delta-31..delta+32
#define NTHREADS 512                // 16 wave32
#define CPT      4                  // x-chunk elems per thread: 2048/512

__global__ __launch_bounds__(NTHREADS)
void longconv_wmma_f16(const float* __restrict__ x,
                       const float* __restrict__ filt,
                       float* __restrict__ out)
{
    // xs (f16 circular stripe, 128KB) and ob (f32 staging, 128KB) are
    // time-disjoint -> one shared LDS block.  fs is a 4-deep ring (8KB).
    __shared__ alignas(32) unsigned char smem[HT * B_ * SBUF * sizeof(_Float16)];
    __shared__ alignas(32) _Float16 fs[4][HT][FW];
    _Float16 (*xs)[B_][SBUF]  = reinterpret_cast<_Float16 (*)[B_][SBUF]>(&smem[0]);
    float    (*ob)[B_][TSPAN] = reinterpret_cast<float (*)[B_][TSPAN]>(&smem[0]);

    const int tid  = threadIdx.x;
    const int wave = tid >> 5;          // channel within tile
    const int lane = tid & 31;
    const int hb   = blockIdx.x * HT;
    const int TB   = blockIdx.y * TSPAN;

    // WMMA fragment coordinates (wave32 layouts, ISA 7.12.2)
    const int n16  = lane & 15;         // A: row M.  B/C/D: column N.
    const int bcol = n16 & 3;           // N -> batch
    const int tt   = n16 >> 2;          // N -> t-subtile
    const int kb_b = (lane >> 4) * 16;  // B: lanes 16-31 hold K=16..31
    const int kb_a = (lane >> 4) * 16;  // A: lanes 16-31 hold K=16..31

    // ---- per-thread staging coordinates (hoisted, live in registers) ----
    int st_hl[CPT], st_bb[CPT], st_sl[CPT];
    const float* st_p[CPT];
    #pragma unroll
    for (int r = 0; r < CPT; ++r) {
        const int e    = tid + r * NTHREADS;
        st_hl[r]       = e & (HT - 1);
        const int rest = e >> 4;                  // [0,128)
        st_sl[r]       = rest & (KW - 1);
        st_bb[r]       = rest >> 5;
        st_p[r] = x + (size_t)st_bb[r] * L_ * H_ + (size_t)st_sl[r] * H_
                    + (size_t)(hb + st_hl[r]);
    }
    int f_hl[2], f_idx[2];
    #pragma unroll
    for (int r = 0; r < 2; ++r) {
        const int e = tid + r * NTHREADS;         // [0,1024) : exactly 2/thread
        f_hl[r]  = e >> 6;
        f_idx[r] = e & (FW - 1);
    }

    // ---- register staging state (2-region pipeline) ----
    float xr[CPT], fr[2];

    // chunks are 32-aligned -> never straddle [0,L): uniform load-or-zero
    auto load_x = [&](int s_base) {
        if (s_base >= 0 && s_base <= L_ - KW) {
            const bool pf = (s_base >= KW);
            #pragma unroll
            for (int r = 0; r < CPT; ++r) {
                const float* p = st_p[r] + (ptrdiff_t)s_base * H_;
                if (pf) __builtin_prefetch(p - (ptrdiff_t)KW * H_, 0, 3);
                xr[r] = *p;
            }
        } else {
            #pragma unroll
            for (int r = 0; r < CPT; ++r) xr[r] = 0.0f;
        }
    };
    auto store_x = [&](int s_base) {
        #pragma unroll
        for (int r = 0; r < CPT; ++r)
            xs[st_hl[r]][st_bb[r]][(s_base + st_sl[r]) & SMASK] = (_Float16)xr[r];
    };
    auto load_f = [&](int dn) {                   // window f = dn-31 .. dn+32
        if (dn >= KW && dn <= L_ - FW) {          // uniform fast path
            #pragma unroll
            for (int r = 0; r < 2; ++r)
                fr[r] = filt[(size_t)(hb + f_hl[r]) * L_ + (dn - 31 + f_idx[r])];
        } else {
            #pragma unroll
            for (int r = 0; r < 2; ++r) {
                const int f = dn - 31 + f_idx[r];
                fr[r] = (f >= 0 && f < L_)
                      ? filt[(size_t)(hb + f_hl[r]) * L_ + f] : 0.0f;
            }
        }
    };
    auto store_f = [&](int ring) {
        #pragma unroll
        for (int r = 0; r < 2; ++r)
            fs[ring][f_hl[r]][f_idx[r]] = (_Float16)fr[r];
    };

    v8f acc[NC];
    #pragma unroll
    for (int c = 0; c < NC; ++c) acc[c] = (v8f){};

    const int ndelta = (TB >> 5) + 2 * NC + 1;    // delta = 0 .. TB+512, step 32

    // ---- prologue: window for delta=0 is s in [TB, TB+528) = chunks -16..0
    for (int m = 2 * NC; m >= 0; --m) { load_x(TB + KW * m); store_x(TB + KW * m); }
    load_f(0); store_f(0);                        // fs ring slot 0 (delta=0)
    load_x(TB - KW);                              // preload stage 1 (chunk it=1)
    load_f(KW);                                   //   and fs window delta=32
    __syncthreads();

    const int lane_sb = TB + 16 * tt + kb_b;      // B-frag s-base (+64c, -delta)
    const int a_base  = 31 + n16 - kb_a;          // fs index for this lane's K=kb_a

    for (int it = 0; it < ndelta; ++it) {
        const int delta = it << 5;
        // store stage it+1 (regs loaded one region earlier); slots are below
        // every slot C(it) reads -> safe with a single barrier per region
        if (it + 1 < ndelta) {
            store_x(TB - ((it + 1) << 5));
            store_f((it + 1) & 3);
        }
        // issue loads for stage it+2 (no wait: hidden behind the 8 WMMAs)
        if (it + 2 < ndelta) {
            load_x(TB - ((it + 2) << 5));
            load_f((it + 2) << 5);
        }

        // A[i][k] = filt[h, delta+i-k] : flipped gather from the window ring
        const int ring = it & 3;
        v16h afrag;
        #pragma unroll
        for (int e = 0; e < 16; ++e) {
            const int off = e;                    // K = kb_a + off
            afrag[e] = fs[ring][wave][a_base - off];
        }
        // one A fragment drives up to NC WMMAs (uniform liveness branches)
        #pragma unroll
        for (int c = 0; c < NC; ++c) {
            if (delta <= TB + TG * c + TG) {
                const int slot = (lane_sb + TG * c - delta) & SMASK; // mult of 16
                const v16h bfrag = *(const v16h*)&xs[wave][bcol][slot];
                acc[c] = __builtin_amdgcn_wmma_f32_16x16x32_f16(
                             false, afrag, false, bfrag, (short)0, acc[c],
                             false, false);
            }
        }
        __syncthreads();
    }

    // ---- stage C tiles (lane=N, vgpr r -> M = r + 8*(lane>=16)) into LDS ----
    {
        const int ih = (lane >> 4) * 8;
        #pragma unroll
        for (int c = 0; c < NC; ++c) {
            #pragma unroll
            for (int r = 0; r < 8; ++r)
                ob[wave][bcol][c * TG + tt * 16 + ih + r] = acc[c][r];
        }
    }
    __syncthreads();

    // ---- coalesced writeback: out = y + u ----
    #pragma unroll 4
    for (int r = 0; r < HT * B_ * TSPAN / NTHREADS; ++r) {   // 64 per thread
        const int e    = tid + r * NTHREADS;
        const int hl   = e & (HT - 1);
        const int rest = e >> 4;                   // [0,2048)
        const int tl   = rest & (TSPAN - 1);
        const int bb   = rest >> 9;
        const size_t gi = ((size_t)bb * L_ + (TB + tl)) * H_ + (hb + hl);
        out[gi] = ob[hl][bb][tl] + x[gi];
    }
}

extern "C" void kernel_launch(void* const* d_in, const int* in_sizes, int n_in,
                              void* d_out, int out_size, void* d_ws, size_t ws_size,
                              hipStream_t stream) {
    const float* x    = (const float*)d_in[0];   // [4, 4096, 1024] f32
    const float* filt = (const float*)d_in[1];   // [1024, 4096] f32
    float* out        = (float*)d_out;           // [4, 4096, 1024] f32
    (void)in_sizes; (void)n_in; (void)out_size; (void)d_ws; (void)ws_size;

    dim3 grid(H_ / HT, L_ / TSPAN);              // 64 x 8 workgroups
    dim3 block(NTHREADS);                        // 16 wave32
    longconv_wmma_f16<<<grid, block, 0, stream>>>(x, filt, out);
}
